// MultiHeadAttention_721554505799
// MI455X (gfx1250) — matrix operand order repaired and checked by reading, not compiled, not run on hardware
//
#include <hip/hip_runtime.h>
#include <hip/hip_bf16.h>

// ---------------------------------------------------------------------------
// MI455X (gfx1250) causal multi-head attention, bf16 WMMA pipeline.
// Compute-bound (~137 GFLOP vs ~80MB HBM traffic) -> all GEMMs through
// v_wmma_f32_16x16x32_bf16. GEMM operand tiles are staged in LDS with the
// CDNA5 async-to-LDS path (global_load_async_to_lds_b128, ASYNCcnt),
// double-buffered so the copy of k-step i+1 overlaps the WMMA burst of i.
// ---------------------------------------------------------------------------

typedef __bf16 bf16_t;
typedef __attribute__((ext_vector_type(16))) __bf16 v16bf;
typedef __attribute__((ext_vector_type(8)))  __bf16 v8bf;
typedef __attribute__((ext_vector_type(8)))  float  v8f;

#define EMBED  1024
#define NHEAD  16
#define DKDIM  64
#define BATCH  4
#define SEQ    2048
#define MTOK   (BATCH * SEQ)     // 8192 tokens
#define NQKV   (3 * EMBED)       // 3072
// 1/sqrt(d_k) folded together with log2(e): softmax done in base-2 (v_exp_f32)
#define QSCALE (0.125f * 1.44269504088896340736f)
#define NEGBIG (-1e30f)

static __device__ __forceinline__ v8f zero8() {
  v8f z;
#pragma unroll
  for (int i = 0; i < 8; ++i) z[i] = 0.0f;
  return z;
}

static __device__ __forceinline__ v8f wmma_bf16(v16bf a, v16bf b, v8f c) {
  // (neg_a, A, neg_b, B, c_mod, C, reuse_a, reuse_b)
  return __builtin_amdgcn_wmma_f32_16x16x32_bf16(false, a, false, b, (short)0, c,
                                                 false, false);
}

// Async DMA of one 16B chunk: global (saddr + voff) -> LDS[lds_off].
// GLOBAL_LOAD_ASYNC_TO_LDS_B128, GVS addressing; tracked by ASYNCcnt.
static __device__ __forceinline__ void async_copy16(unsigned int lds_off,
                                                    unsigned long long gbase,
                                                    int goff_bytes) {
  asm volatile("global_load_async_to_lds_b128 %0, %1, %2"
               :
               : "v"(lds_off), "v"(goff_bytes), "s"(gbase)
               : "memory");
}
static __device__ __forceinline__ void wait_async0() {
  asm volatile("s_wait_asynccnt 0x0" ::: "memory");
}

// A fragment (16x32 = MxK) from row-major [*, ld] at (m0, k0).
// ISA layout: lane L holds row m = L&15; K = (L>>4)*8 + {0..7} and +16.
static __device__ __forceinline__ v16bf load_a(const bf16_t* __restrict__ base,
                                               int ld, int m0, int k0, int lane) {
  const int m  = m0 + (lane & 15);
  const int kb = k0 + ((lane >> 4) << 3);
  const bf16_t* p = base + (size_t)m * ld + kb;
  v8bf lo = *reinterpret_cast<const v8bf*>(p);        // K = kb .. kb+7
  v8bf hi = *reinterpret_cast<const v8bf*>(p + 16);   // K = kb+16 .. kb+23
  return __builtin_shufflevector(lo, hi, 0, 1, 2, 3, 4, 5, 6, 7,
                                 8, 9, 10, 11, 12, 13, 14, 15);
}

// B fragment (32x16 = KxN) read from K-major BT [N][ld] at (k0, n0).
// ISA layout: lane L holds col n = L&15; K = (L>>4)*8 + {0..7} and +16.
static __device__ __forceinline__ v16bf load_b(const bf16_t* __restrict__ baseT,
                                               int ld, int n0, int k0, int lane) {
  const int n  = n0 + (lane & 15);
  const int kb = k0 + ((lane >> 4) << 3);
  const bf16_t* p = baseT + (size_t)n * ld + kb;
  v8bf lo = *reinterpret_cast<const v8bf*>(p);
  v8bf hi = *reinterpret_cast<const v8bf*>(p + 16);
  return __builtin_shufflevector(lo, hi, 0, 1, 2, 3, 4, 5, 6, 7,
                                 8, 9, 10, 11, 12, 13, 14, 15);
}

// Row-wise reductions across the 16 lanes sharing a C-layout row group.
static __device__ __forceinline__ float hmax16(float v) {
#pragma unroll
  for (int m = 1; m < 16; m <<= 1) v = fmaxf(v, __shfl_xor(v, m, 32));
  return v;
}
static __device__ __forceinline__ float hsum16(float v) {
#pragma unroll
  for (int m = 1; m < 16; m <<= 1) v += __shfl_xor(v, m, 32);
  return v;
}

// ---------------------------------------------------------------------------
// Shared GEMM mainloop: C[128x256 block] = A[M,1024] x BT[N,1024]^T.
// 8 waves (2x4) of 64x64 tiles. A/B tiles double-buffered in LDS via
// async-to-LDS DMA. Per k-step: A tile 128x32 (512 x 16B chunks, 2/thread),
// B tile 256x32 (1024 chunks, 4/thread).
// ---------------------------------------------------------------------------
static __device__ __forceinline__ void gemm_mainloop(
    const bf16_t* __restrict__ A, const bf16_t* __restrict__ BT,
    bf16_t* As,   // [2][128*32] LDS
    bf16_t* Bs,   // [2][256*32] LDS
    int m0blk, int n0blk, int tid, v8f acc[4][4]) {
  const int lane = tid & 31;
  const int wave = tid >> 5;
  const int mw = (wave >> 2) * 64;
  const int nw = (wave & 3) * 64;
  const unsigned long long abase = (unsigned long long)(size_t)A;
  const unsigned long long bbase = (unsigned long long)(size_t)BT;

  auto issue = [&](int buf, int k0) {
    const unsigned int a_lds = (unsigned int)(size_t)(As + buf * (128 * 32));
    const unsigned int b_lds = (unsigned int)(size_t)(Bs + buf * (256 * 32));
#pragma unroll
    for (int q = 0; q < 2; ++q) {           // A tile: chunks tid, tid+256
      const int c = tid + 256 * q;
      const int row = c >> 2, col = (c & 3) << 3;   // col in elements
      async_copy16(a_lds + (unsigned int)(row * 32 + col) * 2, abase,
                   ((m0blk + row) * EMBED + k0 + col) * 2);
    }
#pragma unroll
    for (int q = 0; q < 4; ++q) {           // B tile: chunks tid + 256q
      const int c = tid + 256 * q;
      const int row = c >> 2, col = (c & 3) << 3;
      async_copy16(b_lds + (unsigned int)(row * 32 + col) * 2, bbase,
                   ((n0blk + row) * EMBED + k0 + col) * 2);
    }
  };

#pragma unroll
  for (int i = 0; i < 4; ++i)
#pragma unroll
    for (int j = 0; j < 4; ++j) acc[i][j] = zero8();

  issue(0, 0);
  for (int k0 = 0; k0 < EMBED; k0 += 32) {
    const int buf = (k0 >> 5) & 1;
    wait_async0();        // this wave's outstanding DMA done
    __syncthreads();      // every wave's tile landed; prev buffer free
    if (k0 + 32 < EMBED) issue(buf ^ 1, k0 + 32);  // overlap next DMA

    const bf16_t* At = As + buf * (128 * 32);
    const bf16_t* Bt = Bs + buf * (256 * 32);
    v16bf af[4], bfv[4];
#pragma unroll
    for (int i = 0; i < 4; ++i) af[i] = load_a(At, 32, mw + 16 * i, 0, lane);
#pragma unroll
    for (int j = 0; j < 4; ++j) bfv[j] = load_b(Bt, 32, nw + 16 * j, 0, lane);
#pragma unroll
    for (int i = 0; i < 4; ++i)
#pragma unroll
      for (int j = 0; j < 4; ++j) acc[i][j] = wmma_bf16(af[i], bfv[j], acc[i][j]);
  }
}

// ---------------------------------------------------------------------------
// Elementwise conversion kernels
// ---------------------------------------------------------------------------
__global__ void f32_to_bf16_kernel(const float* __restrict__ in,
                                   bf16_t* __restrict__ out, int n) {
  int i = blockIdx.x * 256 + threadIdx.x;
  if (i < n) out[i] = (bf16_t)in[i];
}

// W [K][N] (f32, row-major, y = x@W) -> WT [N][K] (bf16, K-major for B frags)
__global__ void transpose_to_bf16_kernel(const float* __restrict__ W,
                                         bf16_t* __restrict__ WT, int K, int N) {
  int idx = blockIdx.x * 256 + threadIdx.x;
  if (idx >= K * N) return;
  int n = idx / K;
  int k = idx - n * K;
  WT[idx] = (bf16_t)W[(size_t)k * N + n];
}

// ---------------------------------------------------------------------------
// QKV projection GEMM: [8192,1024] x [1024,3072] (+bias), epilogue scatters
// into head-major Q (pre-scaled), K, and transposed V^T layouts.
// ---------------------------------------------------------------------------
__global__ __launch_bounds__(256) void qkv_gemm_kernel(
    const bf16_t* __restrict__ A,   // x_bf16 [8192,1024]
    const bf16_t* __restrict__ BT,  // WqkvT  [3072,1024]
    const float*  __restrict__ bias,
    bf16_t* __restrict__ Qo,        // [B*H, SEQ, 64]
    bf16_t* __restrict__ Ko,        // [B*H, SEQ, 64]
    bf16_t* __restrict__ VTo) {     // [B*H, 64, SEQ]
  __shared__ __align__(16) bf16_t As[2][128 * 32];   //  16 KB
  __shared__ __align__(16) bf16_t Bs[2][256 * 32];   //  32 KB

  const int lane = threadIdx.x & 31;
  const int wave = threadIdx.x >> 5;
  const int m0 = blockIdx.x * 128 + (wave >> 2) * 64;
  const int n0 = blockIdx.y * 256 + (wave & 3) * 64;

  v8f acc[4][4];
  gemm_mainloop(A, BT, &As[0][0], &Bs[0][0],
                blockIdx.x * 128, blockIdx.y * 256, threadIdx.x, acc);

  // Epilogue: C layout lane holds col n0+16j+(lane&15), rows (lane>>4)*8 + r.
#pragma unroll
  for (int i = 0; i < 4; ++i) {
#pragma unroll
    for (int j = 0; j < 4; ++j) {
      const int gn  = n0 + 16 * j + (lane & 15);
      const float bs = bias[gn];
      const int sel = gn >> 10;          // 0=q 1=k 2=v
      const int rem = gn & 1023;
      const int h   = rem >> 6;
      const int d   = rem & 63;
#pragma unroll
      for (int r = 0; r < 8; ++r) {
        const int gm = m0 + 16 * i + ((lane >> 4) << 3) + r;
        const int b  = gm >> 11;
        const int t  = gm & (SEQ - 1);
        const int bh = (b << 4) + h;
        const float v = acc[i][j][r] + bs;
        if (sel == 0) {
          Qo[((size_t)bh * SEQ + t) * DKDIM + d] = (bf16_t)(v * QSCALE);
        } else if (sel == 1) {
          Ko[((size_t)bh * SEQ + t) * DKDIM + d] = (bf16_t)v;
        } else {
          VTo[((size_t)bh * DKDIM + d) * SEQ + t] = (bf16_t)v;
        }
      }
    }
  }
}

// ---------------------------------------------------------------------------
// Causal flash attention. Grid: (SEQ/128, B*H). Block = 256 thr = 8 waves;
// each wave owns 16 query rows, streams 64-key blocks with online softmax.
// ---------------------------------------------------------------------------
__global__ __launch_bounds__(256) void flash_attn_kernel(
    const bf16_t* __restrict__ Q,   // [B*H, SEQ, 64]  (pre-scaled by QSCALE)
    const bf16_t* __restrict__ K,   // [B*H, SEQ, 64]
    const bf16_t* __restrict__ VT,  // [B*H, 64, SEQ]
    bf16_t* __restrict__ Obf) {     // [B*SEQ, EMBED]
  __shared__ bf16_t plds[8][16 * 64];  // per-wave P staging (C->A relayout)

  const int lane = threadIdx.x & 31;
  const int wave = threadIdx.x >> 5;
  const int bh = blockIdx.y;
  const int qw = blockIdx.x * 128 + wave * 16;  // this wave's first query row

  const bf16_t* Qh  = Q  + (size_t)bh * SEQ * DKDIM;
  const bf16_t* Kh  = K  + (size_t)bh * SEQ * DKDIM;
  const bf16_t* VTh = VT + (size_t)bh * DKDIM * SEQ;

  v16bf qf[2];
#pragma unroll
  for (int kk = 0; kk < 2; ++kk) qf[kk] = load_a(Qh, DKDIM, qw, kk * 32, lane);

  v8f accO[4];
#pragma unroll
  for (int n = 0; n < 4; ++n) accO[n] = zero8();
  float m_r[8], l_r[8];
#pragma unroll
  for (int r = 0; r < 8; ++r) { m_r[r] = NEGBIG; l_r[r] = 0.0f; }

  const int rowbase = qw + ((lane >> 4) << 3);
  const int nkb = (qw + 79) >> 6;  // key blocks needed to reach row qw+15

  for (int kb = 0; kb < nkb; ++kb) {
    const int key0 = kb * 64;
    // S = Q K^T : 16x64 per wave
    v8f s[4];
#pragma unroll
    for (int j = 0; j < 4; ++j) {
      s[j] = zero8();
#pragma unroll
      for (int kk = 0; kk < 2; ++kk) {
        v16bf kf = load_b(Kh, DKDIM, key0 + 16 * j, kk * 32, lane);
        s[j] = wmma_bf16(qf[kk], kf, s[j]);
      }
    }
    const bool lastblk = (kb == nkb - 1);  // only diagonal block needs masking
    const int colbase = key0 + (lane & 15);

    bf16_t* myP = &plds[wave][0];
#pragma unroll
    for (int r = 0; r < 8; ++r) {
      float s0 = s[0][r], s1 = s[1][r], s2 = s[2][r], s3 = s[3][r];
      if (lastblk) {
        const int row = rowbase + r;
        if (colbase      > row) s0 = NEGBIG;
        if (colbase + 16 > row) s1 = NEGBIG;
        if (colbase + 32 > row) s2 = NEGBIG;
        if (colbase + 48 > row) s3 = NEGBIG;
      }
      float mx = hmax16(fmaxf(fmaxf(s0, s1), fmaxf(s2, s3)));
      const float mnew = fmaxf(m_r[r], mx);
      const float sc = exp2f(m_r[r] - mnew);
      const float p0 = exp2f(s0 - mnew), p1 = exp2f(s1 - mnew);
      const float p2 = exp2f(s2 - mnew), p3 = exp2f(s3 - mnew);
      const float rs = hsum16(p0 + p1 + p2 + p3);
      l_r[r] = l_r[r] * sc + rs;
      m_r[r] = mnew;
#pragma unroll
      for (int n = 0; n < 4; ++n) accO[n][r] *= sc;
      // C layout -> row-major P tile in LDS
      bf16_t* pr = myP + (((lane >> 4) << 3) + r) * 64 + (lane & 15);
      pr[0] = (bf16_t)p0; pr[16] = (bf16_t)p1;
      pr[32] = (bf16_t)p2; pr[48] = (bf16_t)p3;
    }
    // O += P V  (same-wave DS ops are in-order; no barrier needed)
    v16bf pf[2];
#pragma unroll
    for (int kk = 0; kk < 2; ++kk) pf[kk] = load_a(myP, 64, 0, kk * 32, lane);
#pragma unroll
    for (int n = 0; n < 4; ++n) {
#pragma unroll
      for (int kk = 0; kk < 2; ++kk) {
        v16bf vf = load_b(VTh, SEQ, n * 16, key0 + kk * 32, lane);
        accO[n] = wmma_bf16(pf[kk], vf, accO[n]);
      }
    }
  }

  // Normalize and store to [B*SEQ, EMBED] bf16 for the output projection.
  const int b = bh >> 4, h = bh & 15;
#pragma unroll
  for (int r = 0; r < 8; ++r) {
    const float inv = 1.0f / l_r[r];
    const int t = rowbase + r;
#pragma unroll
    for (int n = 0; n < 4; ++n) {
      const int col = h * 64 + n * 16 + (lane & 15);
      Obf[((size_t)b * SEQ + t) * EMBED + col] = (bf16_t)(accO[n][r] * inv);
    }
  }
}

// ---------------------------------------------------------------------------
// Output projection GEMM: [8192,1024] x [1024,1024] + bias -> f32 d_out.
// ---------------------------------------------------------------------------
__global__ __launch_bounds__(256) void out_gemm_kernel(
    const bf16_t* __restrict__ A,   // Obf [8192,1024]
    const bf16_t* __restrict__ BT,  // WoT [1024,1024]
    const float*  __restrict__ bias,
    float* __restrict__ out) {
  __shared__ __align__(16) bf16_t As[2][128 * 32];
  __shared__ __align__(16) bf16_t Bs[2][256 * 32];

  const int lane = threadIdx.x & 31;
  const int wave = threadIdx.x >> 5;
  const int m0 = blockIdx.x * 128 + (wave >> 2) * 64;
  const int n0 = blockIdx.y * 256 + (wave & 3) * 64;

  v8f acc[4][4];
  gemm_mainloop(A, BT, &As[0][0], &Bs[0][0],
                blockIdx.x * 128, blockIdx.y * 256, threadIdx.x, acc);

#pragma unroll
  for (int i = 0; i < 4; ++i) {
#pragma unroll
    for (int j = 0; j < 4; ++j) {
      const int gn = n0 + 16 * j + (lane & 15);
      const float bs = bias[gn];
#pragma unroll
      for (int r = 0; r < 8; ++r) {
        const int gm = m0 + 16 * i + ((lane >> 4) << 3) + r;
        out[(size_t)gm * EMBED + gn] = acc[i][j][r] + bs;
      }
    }
  }
}

// ---------------------------------------------------------------------------
// Launch
// ---------------------------------------------------------------------------
extern "C" void kernel_launch(void* const* d_in, const int* in_sizes, int n_in,
                              void* d_out, int out_size, void* d_ws, size_t ws_size,
                              hipStream_t stream) {
  (void)in_sizes; (void)n_in; (void)out_size; (void)ws_size;
  const float* x     = (const float*)d_in[0];  // [4,2048,1024]
  const float* W_qkv = (const float*)d_in[1];  // [1024,3072]
  const float* b_qkv = (const float*)d_in[2];  // [3072]
  const float* W_o   = (const float*)d_in[3];  // [1024,1024]
  const float* b_o   = (const float*)d_in[4];  // [1024]
  float* out = (float*)d_out;                  // [4,2048,1024] f32

  // Workspace layout (all 256B-aligned sections), total ~92 MB.
  char* ws = (char*)d_ws;
  size_t off = 0;
  bf16_t* xbf   = (bf16_t*)(ws + off); off += (size_t)MTOK * EMBED * 2;   // 16 MB
  bf16_t* WqkvT = (bf16_t*)(ws + off); off += (size_t)NQKV * EMBED * 2;   //  6 MB
  bf16_t* WoT   = (bf16_t*)(ws + off); off += (size_t)EMBED * EMBED * 2;  //  2 MB
  bf16_t* Qbf   = (bf16_t*)(ws + off); off += (size_t)MTOK * EMBED * 2;   // 16 MB
  bf16_t* Kbf   = (bf16_t*)(ws + off); off += (size_t)MTOK * EMBED * 2;   // 16 MB
  bf16_t* VTbf  = (bf16_t*)(ws + off); off += (size_t)MTOK * EMBED * 2;   // 16 MB
  bf16_t* Obf   = (bf16_t*)(ws + off); off += (size_t)MTOK * EMBED * 2;   // 16 MB

  // 1) precision conversion / weight transposition (bandwidth-trivial)
  f32_to_bf16_kernel<<<(MTOK * EMBED) / 256, 256, 0, stream>>>(x, xbf, MTOK * EMBED);
  transpose_to_bf16_kernel<<<(EMBED * NQKV) / 256, 256, 0, stream>>>(W_qkv, WqkvT, EMBED, NQKV);
  transpose_to_bf16_kernel<<<(EMBED * EMBED) / 256, 256, 0, stream>>>(W_o, WoT, EMBED, EMBED);

  // 2) QKV projection + head split (Q pre-scaled, V transposed)
  qkv_gemm_kernel<<<dim3(MTOK / 128, NQKV / 256), 256, 0, stream>>>(
      xbf, WqkvT, b_qkv, Qbf, Kbf, VTbf);

  // 3) causal flash attention
  flash_attn_kernel<<<dim3(SEQ / 128, BATCH * NHEAD), 256, 0, stream>>>(
      Qbf, Kbf, VTbf, Obf);

  // 4) output projection
  out_gemm_kernel<<<dim3(MTOK / 128, EMBED / 256), 256, 0, stream>>>(
      Obf, WoT, b_o, out);
}
